// Block_17145509446466
// MI455X (gfx1250) — matrix operand order repaired and checked by reading, not compiled, not run on hardware
//
#include <hip/hip_runtime.h>
#include <hip/hip_bf16.h>
#include <stdint.h>

// ---------------------------------------------------------------------------
// Types for CDNA5 (gfx1250) wave32 WMMA
// ---------------------------------------------------------------------------
typedef __attribute__((ext_vector_type(16))) __bf16 v16bf;
typedef __attribute__((ext_vector_type(8)))  __bf16 v8bf;
typedef __attribute__((ext_vector_type(4)))  __bf16 v4bf;
typedef __attribute__((ext_vector_type(8)))  float  v8f;

#define NTOK   65536   // B*T
#define DMODEL 384
#define NHEAD  6
#define HSZ    64
#define TSEQ   256
#define DFF    1536

__device__ __forceinline__ v8f wmma_bf16(v16bf a, v16bf b, v8f c) {
    // (neg_a, A, neg_b, B, c_mod, C, reuse_a, reuse_b)
    return __builtin_amdgcn_wmma_f32_16x16x32_bf16(false, a, false, b,
                                                   (short)0, c, false, false);
}

// gfx1250 async global->LDS copy (16B per lane), tracked by ASYNCcnt.
__device__ __forceinline__ uint32_t lds_off(const void* p) {
    // generic LDS pointer: addr[31:0] is the wave-relative LDS offset
    return (uint32_t)(uintptr_t)p;
}
__device__ __forceinline__ void async_b128(uint32_t lds, const void* gaddr) {
    asm volatile("global_load_async_to_lds_b128 %0, %1, off"
                 :: "v"(lds), "v"(gaddr) : "memory");
}
#define WAIT_ASYNC(n) asm volatile("s_wait_asynccnt " #n ::: "memory")

// A-fragment (16x32 bf16, row-major tile, `pitch` elements per row):
// lane<16 : row = lane,    K = {0..7, 16..23}
// lane>=16: row = lane-16, K = {8..15, 24..31}
__device__ __forceinline__ v16bf frag_a(const __bf16* tile, int pitch) {
    const int lane = threadIdx.x & 31;
    const __bf16* p = tile + (lane & 15) * pitch + ((lane >> 4) << 3);
    v8bf lo = *(const v8bf*)(p);
    v8bf hi = *(const v8bf*)(p + 16);
    v16bf out;
#pragma unroll
    for (int i = 0; i < 8; ++i) { out[i] = lo[i]; out[i + 8] = hi[i]; }
    return out;
}

// B-fragment from a transposed tile Bt[N][K] (pitch elems per row):
// lane<16 : col N = lane,    K = {0..7, 16..23} ; lane>=16: col N = lane-16, K = {8..15, 24..31}
__device__ __forceinline__ v16bf frag_b(const __bf16* tileBt, int pitch) {
    const int lane = threadIdx.x & 31;
    const __bf16* p = tileBt + (lane & 15) * pitch + ((lane >> 4) << 3);
    v8bf lo = *(const v8bf*)(p);
    v8bf hi = *(const v8bf*)(p + 16);
    v16bf out;
#pragma unroll
    for (int i = 0; i < 8; ++i) { out[i] = lo[i]; out[i + 8] = hi[i]; }
    return out;
}

// ---------------------------------------------------------------------------
// Weight conversion kernels: all weights stored TRANSPOSED ([N][K] bf16) so
// GEMM B-staging is a contiguous async copy (no hot-loop transposes).
// ---------------------------------------------------------------------------
__global__ void __launch_bounds__(256) convert_qkv_t_kernel(
        const float* __restrict__ w, __bf16* __restrict__ out) {
    // w: [H][D][HS] fp32 -> out: [H*HS][D] bf16   (N = h*HS+e, K = d)
    int idx = blockIdx.x * 256 + threadIdx.x;
    if (idx >= NHEAD * DMODEL * HSZ) return;
    int e = idx % HSZ;
    int d = (idx / HSZ) % DMODEL;
    int h = idx / (HSZ * DMODEL);
    out[(size_t)(h * HSZ + e) * DMODEL + d] = (__bf16)w[idx];
}

__global__ void __launch_bounds__(256) convert_t_kernel(
        const float* __restrict__ in, __bf16* __restrict__ out, int K, int N) {
    // in: [K][N] fp32 -> out: [N][K] bf16
    int idx = blockIdx.x * 256 + threadIdx.x;
    if (idx >= K * N) return;
    int n = idx % N;
    int kk = idx / N;
    out[(size_t)n * K + kk] = (__bf16)in[idx];
}

// ---------------------------------------------------------------------------
// LayerNorm: one wave per row of 384, shuffle reductions, bf16 output
// ---------------------------------------------------------------------------
__global__ void __launch_bounds__(256) layernorm_kernel(
        const float* __restrict__ x, const float* __restrict__ g,
        const float* __restrict__ be, __bf16* __restrict__ out) {
    const int row  = blockIdx.x * 8 + (threadIdx.x >> 5);
    const int lane = threadIdx.x & 31;
    const float* xr = x + (size_t)row * DMODEL;
    float v[12];
    float s = 0.0f;
#pragma unroll
    for (int i = 0; i < 3; ++i) {
        float4 t = *(const float4*)(xr + i * 128 + lane * 4);
        v[i * 4 + 0] = t.x; v[i * 4 + 1] = t.y;
        v[i * 4 + 2] = t.z; v[i * 4 + 3] = t.w;
    }
#pragma unroll
    for (int i = 0; i < 12; ++i) s += v[i];
#pragma unroll
    for (int m = 1; m < 32; m <<= 1) s += __shfl_xor(s, m, 32);
    const float mu = s * (1.0f / DMODEL);
    float vs = 0.0f;
#pragma unroll
    for (int i = 0; i < 12; ++i) { float d = v[i] - mu; vs += d * d; }
#pragma unroll
    for (int m = 1; m < 32; m <<= 1) vs += __shfl_xor(vs, m, 32);
    const float rs = rsqrtf(vs * (1.0f / DMODEL) + 1e-5f);
    __bf16* orow = out + (size_t)row * DMODEL;
#pragma unroll
    for (int i = 0; i < 3; ++i) {
        v4bf o;
#pragma unroll
        for (int j = 0; j < 4; ++j) {
            int c = i * 128 + lane * 4 + j;
            o[j] = (__bf16)((v[i * 4 + j] - mu) * rs * g[c] + be[c]);
        }
        *(v4bf*)(orow + i * 128 + lane * 4) = o;
    }
}

// ---------------------------------------------------------------------------
// Tiled bf16 WMMA GEMM: C[M,N] = A[M,K] @ Bt[N,K]^T (+bias)(+relu)(+fp32 res)
// BM=128 BN=128 BK=32; 256 thr = 8 waves; wave tile 32x64 (2x4 WMMA).
// Double-buffered LDS tiles filled with async global->LDS copies.
// OUT_VT: write bf16 output in [b][h][e][t] layout (for V) with 16B stores.
// ---------------------------------------------------------------------------
template <bool OUT_BF16, bool DO_BIAS, bool DO_RELU, bool DO_RES, bool OUT_VT>
__global__ void __launch_bounds__(256) gemm_bf16_kernel(
        const __bf16* __restrict__ A, const __bf16* __restrict__ Bt,
        const float* __restrict__ bias, const float* __restrict__ res,
        void* __restrict__ Cout, int M, int N, int K) {
    constexpr int BM = 128, BN = 128, BK = 32, PITCH = BK + 16; // 48
    __shared__ __bf16 As[2][BM * PITCH];   // 2 x 12 KB
    __shared__ __bf16 Bs[2][BN * PITCH];   // 2 x 12 KB

    const int bm = blockIdx.x * BM;
    const int bn = blockIdx.y * BN;
    const int tid  = threadIdx.x;
    const int wid  = tid >> 5;
    const int lane = tid & 31;
    const int wm = (wid >> 1) * 32;   // 4 waves along M
    const int wn = (wid & 1) * 64;    // 2 waves along N
    const int nk = K / BK;

    const int srow = tid >> 2;        // 0..63 staging row
    const int sseg = (tid & 3) * 8;   // 8-half chunk

    auto issue_tile = [&](int ki, int buf) {
        const int k0 = ki * BK;
#pragma unroll
        for (int p = 0; p < 2; ++p) {   // A: 128x32, 16B per lane, 2 rounds
            int row = p * 64 + srow;
            async_b128(lds_off(&As[buf][row * PITCH + sseg]),
                       A + (size_t)(bm + row) * K + k0 + sseg);
        }
#pragma unroll
        for (int p = 0; p < 2; ++p) {   // Bt: 128x32
            int row = p * 64 + srow;
            async_b128(lds_off(&Bs[buf][row * PITCH + sseg]),
                       Bt + (size_t)(bn + row) * K + k0 + sseg);
        }
    };

    v8f acc[2][4] = {};
    issue_tile(0, 0);

    for (int ki = 0; ki < nk; ++ki) {
        const int cur = ki & 1;
        if (ki + 1 < nk) {
            issue_tile(ki + 1, cur ^ 1);
            WAIT_ASYNC(0x4);            // tile ki's 4 copies retired (in-order)
        } else {
            WAIT_ASYNC(0x0);
        }
        __syncthreads();
#pragma unroll
        for (int mi = 0; mi < 2; ++mi) {
            v16bf af = frag_a(&As[cur][(wm + mi * 16) * PITCH], PITCH);
#pragma unroll
            for (int ni = 0; ni < 4; ++ni) {
                v16bf bf_ = frag_b(&Bs[cur][(wn + ni * 16) * PITCH], PITCH);
                acc[mi][ni] = wmma_bf16(af, bf_, acc[mi][ni]);
            }
        }
        __syncthreads();
    }

    if (OUT_VT) {
        // V output: vt[((b*H + h)*HS + e)][t], 8 consecutive t per lane -> 16B
#pragma unroll
        for (int mi = 0; mi < 2; ++mi) {
            const int tg = bm + wm + mi * 16 + 8 * (lane >> 4);
            const int bb = tg >> 8;       // batch index (run of 8 never crosses)
            const int tl = tg & 255;
#pragma unroll
            for (int ni = 0; ni < 4; ++ni) {
                const int col = bn + wn + ni * 16 + (lane & 15); // h*HS+e
                v8bf o;
#pragma unroll
                for (int r = 0; r < 8; ++r) o[r] = (__bf16)acc[mi][ni][r];
                *(v8bf*)((__bf16*)Cout + ((size_t)bb * DMODEL + col) * TSEQ + tl) = o;
            }
        }
    } else {
#pragma unroll
        for (int mi = 0; mi < 2; ++mi) {
#pragma unroll
            for (int ni = 0; ni < 4; ++ni) {
                const int col = bn + wn + ni * 16 + (lane & 15);
                float bi = 0.0f;
                if (DO_BIAS) bi = bias[col];
#pragma unroll
                for (int r = 0; r < 8; ++r) {
                    const int row = bm + wm + mi * 16 + r + 8 * (lane >> 4);
                    float val = acc[mi][ni][r] + bi;
                    if (DO_RELU) val = fmaxf(val, 0.0f);
                    if (DO_RES)  val += res[(size_t)row * N + col];
                    if (OUT_BF16)
                        ((__bf16*)Cout)[(size_t)row * N + col] = (__bf16)val;
                    else
                        ((float*)Cout)[(size_t)row * N + col] = val;
                }
            }
        }
    }
}

// ---------------------------------------------------------------------------
// Flash-style attention, softmax over all 256 keys, scale = 1/sqrt(384).
// block = 128 thr (4 waves); wave -> 16 query rows; 64-key chunks staged
// in double-buffered LDS via async copies. V pre-transposed ([b][h][e][t]).
// grid = (T/64, H, B)
// ---------------------------------------------------------------------------
__global__ void __launch_bounds__(128) attention_kernel(
        const __bf16* __restrict__ q, const __bf16* __restrict__ k,
        const __bf16* __restrict__ vt, __bf16* __restrict__ attn) {
    constexpr int CH = 64;        // key chunk
    constexpr int KP = HSZ + 8;   // 72 (144B rows, 16B-aligned chunks)
    constexpr int PP = 48;        // P-tile pitch
    __shared__ __bf16 Ks[2][CH * KP];    // [s][e]  (Bt for Q@K^T)
    __shared__ __bf16 Vts[2][HSZ * KP];  // [e][s]  (Bt for P@V)
    __shared__ __bf16 Ps[4 * 16 * PP];   // per-wave 16x32 P tile

    const int b  = blockIdx.z;
    const int h  = blockIdx.y;
    const int tid  = threadIdx.x;
    const int wid  = tid >> 5;
    const int lane = tid & 31;
    const int t0 = blockIdx.x * 64 + wid * 16;
    const float scale = 0.05103103630798288f;  // 1/sqrt(384)

    // Q A-fragments, K-dim = HS = 64 -> two 32-wide k-steps
    const __bf16* qrow = q + (size_t)(b * TSEQ + t0 + (lane & 15)) * DMODEL
                           + h * HSZ + ((lane >> 4) << 3);
    v16bf qa[2];
    {
        v8bf l0 = *(const v8bf*)(qrow);
        v8bf h0 = *(const v8bf*)(qrow + 16);
        v8bf l1 = *(const v8bf*)(qrow + 32);
        v8bf h1 = *(const v8bf*)(qrow + 48);
#pragma unroll
        for (int i = 0; i < 8; ++i) {
            qa[0][i] = l0[i]; qa[0][i + 8] = h0[i];
            qa[1][i] = l1[i]; qa[1][i + 8] = h1[i];
        }
    }

    const int srow = tid >> 1;        // 0..63
    const int sseg = (tid & 1) * 32;  // half-row segment
    auto issue_chunk = [&](int j0, int buf) {
        // K chunk: Ks[s][e] from row-major k
        const __bf16* kg = k + (size_t)(b * TSEQ + j0 + srow) * DMODEL + h * HSZ + sseg;
#pragma unroll
        for (int c = 0; c < 4; ++c)
            async_b128(lds_off(&Ks[buf][srow * KP + sseg + c * 8]), kg + c * 8);
        // V^T chunk: Vts[e][s] from vt[((b*H+h)*HS + e)][t]
        const __bf16* vg = vt + ((size_t)(b * NHEAD + h) * HSZ + srow) * TSEQ + j0 + sseg;
#pragma unroll
        for (int c = 0; c < 4; ++c)
            async_b128(lds_off(&Vts[buf][srow * KP + sseg + c * 8]), vg + c * 8);
    };

    float m_run[8], l_run[8];
    v8f acc_o[4] = {};
#pragma unroll
    for (int r = 0; r < 8; ++r) { m_run[r] = -3.0e38f; l_run[r] = 0.0f; }

    __bf16* Pw = &Ps[wid * 16 * PP];

    issue_chunk(0, 0);
    for (int c0 = 0; c0 < TSEQ / CH; ++c0) {
        const int cur = c0 & 1;
        if (c0 + 1 < TSEQ / CH) {
            issue_chunk((c0 + 1) * CH, cur ^ 1);
            WAIT_ASYNC(0x8);          // chunk c0's 8 copies retired
        } else {
            WAIT_ASYNC(0x0);
        }
        __syncthreads();

#pragma unroll
        for (int jj = 0; jj < CH; jj += 32) {
            // S(16x32) = Q @ K^T
            v8f sacc[2] = {};
#pragma unroll
            for (int nt = 0; nt < 2; ++nt) {
                sacc[nt] = wmma_bf16(qa[0], frag_b(&Ks[cur][(jj + nt * 16) * KP + 0],  KP), sacc[nt]);
                sacc[nt] = wmma_bf16(qa[1], frag_b(&Ks[cur][(jj + nt * 16) * KP + 32], KP), sacc[nt]);
            }
            // online softmax (rows: lanes<16 -> r, lanes>=16 -> r+8)
#pragma unroll
            for (int r = 0; r < 8; ++r) {
                float s0 = sacc[0][r] * scale;
                float s1 = sacc[1][r] * scale;
                float rmax = fmaxf(s0, s1);
#pragma unroll
                for (int m = 1; m < 16; m <<= 1) rmax = fmaxf(rmax, __shfl_xor(rmax, m, 32));
                float mnew = fmaxf(m_run[r], rmax);
                float corr = __expf(m_run[r] - mnew);
                float p0 = __expf(s0 - mnew);
                float p1 = __expf(s1 - mnew);
                float rsum = p0 + p1;
#pragma unroll
                for (int m = 1; m < 16; m <<= 1) rsum += __shfl_xor(rsum, m, 32);
                l_run[r] = l_run[r] * corr + rsum;
                m_run[r] = mnew;
#pragma unroll
                for (int e = 0; e < 4; ++e) acc_o[e][r] *= corr;
                const int mrow = r + 8 * (lane >> 4);
                Pw[mrow * PP + (lane & 15)]      = (__bf16)p0;
                Pw[mrow * PP + 16 + (lane & 15)] = (__bf16)p1;
            }
            // O += P(16x32) @ V(32xHS)
            v16bf pf = frag_a(Pw, PP);
#pragma unroll
            for (int e = 0; e < 4; ++e)
                acc_o[e] = wmma_bf16(pf, frag_b(&Vts[cur][(e * 16) * KP + jj], KP), acc_o[e]);
        }
        __syncthreads();
    }

    // normalize and store (bf16, heads concatenated along feature dim)
#pragma unroll
    for (int r = 0; r < 8; ++r) {
        const int row = t0 + r + 8 * (lane >> 4);
        const float invl = 1.0f / l_run[r];
        __bf16* orow = attn + (size_t)(b * TSEQ + row) * DMODEL + h * HSZ;
#pragma unroll
        for (int e = 0; e < 4; ++e)
            orow[e * 16 + (lane & 15)] = (__bf16)(acc_o[e][r] * invl);
    }
}

// ---------------------------------------------------------------------------
// Host-side orchestration
// ---------------------------------------------------------------------------
extern "C" void kernel_launch(void* const* d_in, const int* in_sizes, int n_in,
                              void* d_out, int out_size, void* d_ws, size_t ws_size,
                              hipStream_t stream) {
    const float* x      = (const float*)d_in[0];
    const float* wq     = (const float*)d_in[1];
    const float* wk     = (const float*)d_in[2];
    const float* wv     = (const float*)d_in[3];
    const float* w_proj = (const float*)d_in[4];
    const float* b_proj = (const float*)d_in[5];
    const float* w1     = (const float*)d_in[6];
    const float* b1     = (const float*)d_in[7];
    const float* w2     = (const float*)d_in[8];
    const float* b2     = (const float*)d_in[9];
    const float* g1     = (const float*)d_in[10];
    const float* be1    = (const float*)d_in[11];
    const float* g2     = (const float*)d_in[12];
    const float* be2    = (const float*)d_in[13];
    float* out = (float*)d_out;

    char* ws = (char*)d_ws;
    size_t off = 0;
    auto alloc = [&](size_t bytes) -> void* {
        void* p = ws + off;
        off += (bytes + 255) & ~(size_t)255;
        return p;
    };

    __bf16* wq_t  = (__bf16*)alloc((size_t)DMODEL * DMODEL * 2);
    __bf16* wk_t  = (__bf16*)alloc((size_t)DMODEL * DMODEL * 2);
    __bf16* wv_t  = (__bf16*)alloc((size_t)DMODEL * DMODEL * 2);
    __bf16* wp_t  = (__bf16*)alloc((size_t)DMODEL * DMODEL * 2);
    __bf16* w1_t  = (__bf16*)alloc((size_t)DMODEL * DFF * 2);
    __bf16* w2_t  = (__bf16*)alloc((size_t)DFF * DMODEL * 2);
    __bf16* ln_b  = (__bf16*)alloc((size_t)NTOK * DMODEL * 2);
    __bf16* q_b   = (__bf16*)alloc((size_t)NTOK * DMODEL * 2);
    __bf16* k_b   = (__bf16*)alloc((size_t)NTOK * DMODEL * 2);
    __bf16* vt_b  = (__bf16*)alloc((size_t)NTOK * DMODEL * 2);  // [b][h][e][t]
    __bf16* at_b  = (__bf16*)alloc((size_t)NTOK * DMODEL * 2);
    float*  x1    = (float*) alloc((size_t)NTOK * DMODEL * 4);
    __bf16* h1_b  = (__bf16*)alloc((size_t)NTOK * DFF * 2);
    (void)ws_size; (void)in_sizes; (void)n_in; (void)out_size;

    // weight conversions to transposed bf16 ([N][K])
    convert_qkv_t_kernel<<<576, 256, 0, stream>>>(wq, wq_t);
    convert_qkv_t_kernel<<<576, 256, 0, stream>>>(wk, wk_t);
    convert_qkv_t_kernel<<<576, 256, 0, stream>>>(wv, wv_t);
    convert_t_kernel<<<576, 256, 0, stream>>>(w_proj, wp_t, DMODEL, DMODEL);
    convert_t_kernel<<<2304, 256, 0, stream>>>(w1, w1_t, DMODEL, DFF);
    convert_t_kernel<<<2304, 256, 0, stream>>>(w2, w2_t, DFF, DMODEL);

    const dim3 blk256(256);
    const dim3 gN384(NTOK / 128, DMODEL / 128);  // (512, 3)
    const dim3 gN1536(NTOK / 128, DFF / 128);    // (512, 12)

    // ln1 = LayerNorm(x)
    layernorm_kernel<<<NTOK / 8, blk256, 0, stream>>>(x, g1, be1, ln_b);

    // q/k projections (bf16, row-major), v projection (bf16, [b][h][e][t])
    gemm_bf16_kernel<true, false, false, false, false><<<gN384, blk256, 0, stream>>>(
        ln_b, wq_t, nullptr, nullptr, q_b, NTOK, DMODEL, DMODEL);
    gemm_bf16_kernel<true, false, false, false, false><<<gN384, blk256, 0, stream>>>(
        ln_b, wk_t, nullptr, nullptr, k_b, NTOK, DMODEL, DMODEL);
    gemm_bf16_kernel<true, false, false, false, true><<<gN384, blk256, 0, stream>>>(
        ln_b, wv_t, nullptr, nullptr, vt_b, NTOK, DMODEL, DMODEL);

    // attention (softmax over all keys, scale 1/sqrt(384))
    attention_kernel<<<dim3(TSEQ / 64, NHEAD, 256), dim3(128), 0, stream>>>(
        q_b, k_b, vt_b, at_b);

    // x1 = x + attn @ w_proj + b_proj  (fp32)
    gemm_bf16_kernel<false, true, false, true, false><<<gN384, blk256, 0, stream>>>(
        at_b, wp_t, b_proj, x, x1, NTOK, DMODEL, DMODEL);

    // ln2 = LayerNorm(x1)
    layernorm_kernel<<<NTOK / 8, blk256, 0, stream>>>(x1, g2, be2, ln_b);

    // h1 = relu(ln2 @ w1 + b1)  (bf16)
    gemm_bf16_kernel<true, true, true, false, false><<<gN1536, blk256, 0, stream>>>(
        ln_b, w1_t, b1, nullptr, h1_b, NTOK, DFF, DMODEL);

    // out = x1 + h1 @ w2 + b2  (fp32)
    gemm_bf16_kernel<false, true, false, true, false><<<gN384, blk256, 0, stream>>>(
        h1_b, w2_t, b2, x1, out, NTOK, DMODEL, DFF);
}
